// GPT_50130858279052
// MI455X (gfx1250) — compile-verified, matrix operand order checked by reading, not run on hardware
//
#include <hip/hip_runtime.h>
#include <hip/hip_bf16.h>
#include <math.h>

// GPT-2 124M forward for gfx1250 (MI455X): bf16 WMMA GEMMs + flash-style WMMA attention.
// GEMM pipelines A-tiles through GLOBAL_LOAD_ASYNC_TO_LDS (ASYNCcnt) when available.
typedef __bf16 bf16;
typedef __attribute__((ext_vector_type(16))) __bf16 bf16x16;
typedef __attribute__((ext_vector_type(8)))  float  f32x8;

#define TT   1024
#define DD   768
#define HH   12
#define HDD  64
#define BB   4
#define VV   50257
#define NROWS 4096   // B*T
#define NLAYER 12

#if defined(__gfx1250__) && __has_builtin(__builtin_amdgcn_global_load_async_to_lds_b128) && \
    __has_builtin(__builtin_amdgcn_s_wait_asynccnt)
#define HAS_ASYNC_LDS 1
typedef int v4i __attribute__((vector_size(4 * sizeof(int))));
typedef v4i __attribute__((address_space(1))) g_v4i;
typedef v4i __attribute__((address_space(3))) l_v4i;
__device__ __forceinline__ void async_b128(const void* g, void* l) {
  // (global src v4i AS1*, LDS dst v4i AS3*, imm offset, imm cpol)
  __builtin_amdgcn_global_load_async_to_lds_b128((g_v4i*)g, (l_v4i*)l, 0, 0);
}
#else
#define HAS_ASYNC_LDS 0
#endif

// Column of the element pair (2v, 2v+1) in the 16-bit A/B fragment layout:
// K(e) = ((e&8)<<1) + (e&7) + khalf*8  (ISA 7.12.2, 16-bit A-matrix 16x32)
__device__ __forceinline__ int kpair_col(int v, int khalf) {
  return ((v & 4) << 2) + ((v & 3) << 1) + khalf * 8;
}

union AFrag { bf16x16 v; unsigned u[8]; };
union Pack2 { bf16 h[2]; unsigned u; };
union Pack4 { bf16 h[4]; unsigned long long u; };

__device__ __forceinline__ f32x8 wmma_bf16(const AFrag& a, const AFrag& b, f32x8 c) {
  return __builtin_amdgcn_wmma_f32_16x16x32_bf16(false, a.v, false, b.v, (short)0, c, false, false);
}

// ---------------------------------------------------------------- embedding
__global__ __launch_bounds__(256) void embed_kernel(const int* __restrict__ idx,
    const float* __restrict__ wte, const float* __restrict__ wpe,
    float* __restrict__ x) {
  const int row = blockIdx.x;            // 0..4095
  const int t = row % TT;
  const int id = idx[row];
  const float* we = wte + (size_t)id * DD;
  const float* pe = wpe + (size_t)t * DD;
  float* xr = x + (size_t)row * DD;
  #pragma unroll
  for (int i = 0; i < 3; ++i) {
    const int c = threadIdx.x + i * 256;
    xr[c] = we[c] + pe[c];
  }
}

// ---------------------------------------------------------------- layernorm (fp32 in -> bf16 out)
__global__ __launch_bounds__(256) void ln_kernel(const float* __restrict__ x,
    const float* __restrict__ w, const float* __restrict__ b,
    bf16* __restrict__ out) {
  __shared__ float red[256];
  const int row = blockIdx.x, tid = threadIdx.x;
  const float* xr = x + (size_t)row * DD;
  float s = 0.f;
  #pragma unroll
  for (int i = 0; i < 3; ++i) s += xr[tid + i * 256];
  red[tid] = s; __syncthreads();
  for (int off = 128; off > 0; off >>= 1) {
    if (tid < off) red[tid] += red[tid + off];
    __syncthreads();
  }
  const float mu = red[0] * (1.0f / DD);
  __syncthreads();
  float v = 0.f;
  #pragma unroll
  for (int i = 0; i < 3; ++i) { const float d = xr[tid + i * 256] - mu; v += d * d; }
  red[tid] = v; __syncthreads();
  for (int off = 128; off > 0; off >>= 1) {
    if (tid < off) red[tid] += red[tid + off];
    __syncthreads();
  }
  const float rs = rsqrtf(red[0] * (1.0f / DD) + 1e-5f);
  bf16* orow = out + (size_t)row * DD;
  #pragma unroll
  for (int i = 0; i < 3; ++i) {
    const int c = tid + i * 256;
    orow[c] = (bf16)((xr[c] - mu) * rs * w[c] + b[c]);
  }
}

// ---------------------------------------------------------------- WMMA GEMM
// C[M,N] = A[M,K](bf16) @ B(fp32, converted to bf16 in LDS) + bias, epilogue EPI:
//   0: store bf16   1: GELU(exact) -> bf16   2: fp32 residual add into Cout   3: store fp32
// BT=false: B is [K,N] row-major.  BT=true: B is [N,K] row-major (lm_head: x @ W^T).
template<int EPI, bool BT>
__global__ __launch_bounds__(256) void gemm_kernel(const bf16* __restrict__ A,
    const float* __restrict__ Bw, const float* __restrict__ bias,
    void* __restrict__ Cout, int M, int N, int K) {
  __shared__ __align__(16) bf16 As[2][128][40];   // [m][k], padded stride 40 (80B rows)
  __shared__ __align__(16) bf16 Bs[2][128][40];   // [n][k]
  const int tid  = threadIdx.x;
  const int lane = tid & 31, wid = tid >> 5;
  const int wr = wid & 3, wc = wid >> 2;          // 4x2 wave grid -> 128x128 block tile
  const int m0 = blockIdx.y * 128, n0 = blockIdx.x * 128;
  const int mloc = lane & 15, khalf = lane >> 4;

  const f32x8 fz = {0.f,0.f,0.f,0.f,0.f,0.f,0.f,0.f};
  f32x8 acc[2][4];
  #pragma unroll
  for (int i = 0; i < 2; ++i)
    #pragma unroll
    for (int j = 0; j < 4; ++j) acc[i][j] = fz;

  float4 breg[4];
  auto loadB = [&](int k0) {                      // global fp32 tile -> registers
    #pragma unroll
    for (int i = 0; i < 4; ++i) {
      const int id = tid + i * 256;
      if (BT) {
        const int gn = id >> 3, c4 = id & 7;
        const int n = n0 + gn;
        if (n < N) breg[i] = *(const float4*)(Bw + (size_t)n * K + k0 + c4 * 4);
        else { breg[i].x = breg[i].y = breg[i].z = breg[i].w = 0.f; }
      } else {
        const int gk = id >> 5, c4 = id & 31;
        breg[i] = *(const float4*)(Bw + (size_t)(k0 + gk) * N + n0 + c4 * 4);
      }
    }
  };
  auto storeB = [&](int buf) {                    // registers -> bf16 LDS tile [n][k]
    #pragma unroll
    for (int i = 0; i < 4; ++i) {
      const int id = tid + i * 256;
      if (BT) {
        const int gn = id >> 3, c4 = id & 7;
        Pack4 p; p.h[0]=(bf16)breg[i].x; p.h[1]=(bf16)breg[i].y;
                 p.h[2]=(bf16)breg[i].z; p.h[3]=(bf16)breg[i].w;
        *(unsigned long long*)&Bs[buf][gn][c4 * 4] = p.u;
      } else {
        const int gk = id >> 5, c4 = id & 31;
        Bs[buf][c4 * 4 + 0][gk] = (bf16)breg[i].x;
        Bs[buf][c4 * 4 + 1][gk] = (bf16)breg[i].y;
        Bs[buf][c4 * 4 + 2][gk] = (bf16)breg[i].z;
        Bs[buf][c4 * 4 + 3][gk] = (bf16)breg[i].w;
      }
    }
  };
  auto compute = [&](int buf) {                   // fragments + 8 WMMAs per wave
    AFrag afr[2], bfr[4];
    #pragma unroll
    for (int mi = 0; mi < 2; ++mi)
      #pragma unroll
      for (int v = 0; v < 8; ++v)
        afr[mi].u[v] = *(const unsigned*)&As[buf][wr * 32 + mi * 16 + mloc][kpair_col(v, khalf)];
    #pragma unroll
    for (int ni = 0; ni < 4; ++ni)
      #pragma unroll
      for (int v = 0; v < 8; ++v)
        bfr[ni].u[v] = *(const unsigned*)&Bs[buf][wc * 64 + ni * 16 + mloc][kpair_col(v, khalf)];
    #pragma unroll
    for (int mi = 0; mi < 2; ++mi)
      #pragma unroll
      for (int ni = 0; ni < 4; ++ni)
        acc[mi][ni] = wmma_bf16(afr[mi], bfr[ni], acc[mi][ni]);
  };

  const int ntiles = K >> 5;

#if HAS_ASYNC_LDS
  // ---- pipelined: A tile via async global->LDS DMA (ASYNCcnt), B tile via regs
  auto issueA = [&](int buf, int k0) {
    #pragma unroll
    for (int i = 0; i < 2; ++i) {
      const int id = tid + i * 256;
      const int r = id >> 2, c4 = id & 3;
      async_b128(A + (size_t)(m0 + r) * K + k0 + c4 * 8, &As[buf][r][c4 * 8]);
    }
  };
  issueA(0, 0);
  loadB(0);
  storeB(0);
  for (int kt = 0; kt < ntiles; ++kt) {
    const int cur = kt & 1;
    __builtin_amdgcn_s_wait_asynccnt(0);   // this wave's A(cur) resident in LDS
    __syncthreads();                        // all waves' A(cur) + B(cur) stores visible
    if (kt + 1 < ntiles) {
      issueA(cur ^ 1, (kt + 1) << 5);       // DMA next A tile, overlapped with WMMAs
      loadB((kt + 1) << 5);                 // next B tile global loads in flight
    }
    compute(cur);
    if (kt + 1 < ntiles) storeB(cur ^ 1);
  }
#else
  // ---- fallback: synchronous single-buffer staging (round-1 path)
  for (int kt = 0; kt < ntiles; ++kt) {
    const int k0 = kt << 5;
    #pragma unroll
    for (int i = 0; i < 2; ++i) {
      const int id = tid + i * 256;
      const int r = id >> 2, c4 = id & 3;
      *(uint4*)&As[0][r][c4 * 8] = *(const uint4*)(A + (size_t)(m0 + r) * K + k0 + c4 * 8);
    }
    loadB(k0);
    storeB(0);
    if (k0 + 32 < K)  // prefetch next A tile slice -> global_prefetch_b8
      __builtin_prefetch(A + (size_t)(m0 + (tid >> 1)) * K + (k0 + 32) + (tid & 1) * 16, 0, 1);
    __syncthreads();
    compute(0);
    __syncthreads();
  }
#endif

  // ---- epilogue: C element (M,N): row = r + khalf*8 within subtile, col = mloc
  #pragma unroll
  for (int mi = 0; mi < 2; ++mi) {
    #pragma unroll
    for (int ni = 0; ni < 4; ++ni) {
      #pragma unroll
      for (int r = 0; r < 8; ++r) {
        const int row = m0 + wr * 32 + mi * 16 + r + khalf * 8;
        const int col = n0 + wc * 64 + ni * 16 + mloc;
        if (col < N) {
          float v = acc[mi][ni][r];
          if (bias) v += bias[col];
          const size_t o = (size_t)row * N + col;
          if (EPI == 0) {
            ((bf16*)Cout)[o] = (bf16)v;
          } else if (EPI == 1) {
            const float g = 0.5f * v * (1.0f + erff(v * 0.70710678118654752f));
            ((bf16*)Cout)[o] = (bf16)g;
          } else if (EPI == 2) {
            float* cf = (float*)Cout;
            cf[o] = cf[o] + v;
          } else {
            ((float*)Cout)[o] = v;
          }
        }
      }
    }
  }
}

// ---------------------------------------------------------------- flash attention (one wave / 16-query tile)
__global__ __launch_bounds__(32) void attn_kernel(const bf16* __restrict__ qkv,
                                                  bf16* __restrict__ y) {
  const int lane = threadIdx.x;
  const int mloc = lane & 15, khalf = lane >> 4;
  const int tile = blockIdx.x, h = blockIdx.y, b = blockIdx.z;
  const int q0 = tile * 16;
  const bf16* base = qkv + (size_t)b * TT * (3 * DD);
  __shared__ __align__(8) bf16 P[16][32];   // C-layout -> A-layout transpose staging

  // Q fragments (16 queries x 64 dims = two k32 A-fragments)
  AFrag aq[2];
  {
    const bf16* qptr = base + (size_t)(q0 + mloc) * (3 * DD) + h * HDD;
    #pragma unroll
    for (int f = 0; f < 2; ++f)
      #pragma unroll
      for (int v = 0; v < 8; ++v)
        aq[f].u[v] = *(const unsigned*)(qptr + f * 32 + kpair_col(v, khalf));
  }

  const f32x8 fz = {0.f,0.f,0.f,0.f,0.f,0.f,0.f,0.f};
  f32x8 o[4]; // 16 queries x 64 dims accumulator
  #pragma unroll
  for (int j = 0; j < 4; ++j) o[j] = fz;
  float mi[8], li[8];
  #pragma unroll
  for (int r = 0; r < 8; ++r) { mi[r] = -1e30f; li[r] = 0.f; }

  const int nkb = (q0 + 15) / 32 + 1;       // causal: key blocks of 32 up to last query
  for (int kb = 0; kb < nkb; ++kb) {
    const int ks = kb * 32;
    // ---- S = Q K^T  (two 16x16 key subtiles, each K=64 via 2 WMMAs)
    f32x8 s[2];
    #pragma unroll
    for (int ns = 0; ns < 2; ++ns) {
      const bf16* kptr = base + (size_t)(ks + ns * 16 + mloc) * (3 * DD) + DD + h * HDD;
      AFrag bk0, bk1;
      #pragma unroll
      for (int v = 0; v < 8; ++v) bk0.u[v] = *(const unsigned*)(kptr + kpair_col(v, khalf));
      #pragma unroll
      for (int v = 0; v < 8; ++v) bk1.u[v] = *(const unsigned*)(kptr + 32 + kpair_col(v, khalf));
      f32x8 c = fz;
      c = wmma_bf16(aq[0], bk0, c);
      c = wmma_bf16(aq[1], bk1, c);
      s[ns] = c;
    }
    // ---- scale + causal mask
    #pragma unroll
    for (int ns = 0; ns < 2; ++ns)
      #pragma unroll
      for (int r = 0; r < 8; ++r) {
        const int key = ks + ns * 16 + mloc;
        const int qr  = q0 + r + khalf * 8;
        const float v = s[ns][r] * 0.125f;   // 1/sqrt(64)
        s[ns][r] = (key > qr) ? -1e30f : v;
      }
    // ---- online softmax; row M lives in one VGPR across its 16-lane group
    #pragma unroll
    for (int r = 0; r < 8; ++r) {
      float rm = fmaxf(s[0][r], s[1][r]);
      #pragma unroll
      for (int off = 1; off < 16; off <<= 1) rm = fmaxf(rm, __shfl_xor(rm, off, 32));
      const float mnew  = fmaxf(mi[r], rm);
      const float alpha = __expf(mi[r] - mnew);
      const float p0 = __expf(s[0][r] - mnew);
      const float p1 = __expf(s[1][r] - mnew);
      s[0][r] = p0; s[1][r] = p1;
      float sum = p0 + p1;
      #pragma unroll
      for (int off = 1; off < 16; off <<= 1) sum += __shfl_xor(sum, off, 32);
      li[r] = li[r] * alpha + sum;
      mi[r] = mnew;
      #pragma unroll
      for (int j = 0; j < 4; ++j) o[j][r] *= alpha;
    }
    // ---- P: C-layout -> A-layout via LDS
    #pragma unroll
    for (int ns = 0; ns < 2; ++ns)
      #pragma unroll
      for (int r = 0; r < 8; ++r)
        P[r + khalf * 8][ns * 16 + mloc] = (bf16)s[ns][r];
    __syncthreads();
    AFrag pa;
    #pragma unroll
    for (int v = 0; v < 8; ++v)
      pa.u[v] = *(const unsigned*)&P[mloc][kpair_col(v, khalf)];
    // ---- O += P V  (4 dim-subtiles of 16)
    #pragma unroll
    for (int j = 0; j < 4; ++j) {
      AFrag vb;
      #pragma unroll
      for (int v = 0; v < 8; ++v) {
        const int kk = kpair_col(v, khalf);
        const bf16* v0 = base + (size_t)(ks + kk)     * (3 * DD) + 2 * DD + h * HDD + j * 16 + mloc;
        const bf16* v1 = base + (size_t)(ks + kk + 1) * (3 * DD) + 2 * DD + h * HDD + j * 16 + mloc;
        Pack2 p; p.h[0] = *v0; p.h[1] = *v1;
        vb.u[v] = p.u;
      }
      o[j] = wmma_bf16(pa, vb, o[j]);
    }
    __syncthreads();
  }
  // ---- normalize + store y (bf16, [B*T, D])
  #pragma unroll
  for (int r = 0; r < 8; ++r) {
    const float inv = 1.0f / li[r];
    #pragma unroll
    for (int j = 0; j < 4; ++j) {
      const int row = b * TT + q0 + r + khalf * 8;
      const int col = h * HDD + j * 16 + mloc;
      y[(size_t)row * DD + col] = (bf16)(o[j][r] * inv);
    }
  }
}

// ---------------------------------------------------------------- launcher
extern "C" void kernel_launch(void* const* d_in, const int* in_sizes, int n_in,
                              void* d_out, int out_size, void* d_ws, size_t ws_size,
                              hipStream_t stream) {
  const int*   idx       = (const int*)  d_in[0];
  const float* wte       = (const float*)d_in[1];
  const float* wpe       = (const float*)d_in[2];
  const float* ln1_w     = (const float*)d_in[3];
  const float* ln1_b     = (const float*)d_in[4];
  const float* c_attn_w  = (const float*)d_in[5];
  const float* c_attn_b  = (const float*)d_in[6];
  const float* c_proj_w  = (const float*)d_in[7];
  const float* c_proj_b  = (const float*)d_in[8];
  const float* ln2_w     = (const float*)d_in[9];
  const float* ln2_b     = (const float*)d_in[10];
  const float* fc_w      = (const float*)d_in[11];
  const float* fc_b      = (const float*)d_in[12];
  const float* fc_proj_w = (const float*)d_in[13];
  const float* fc_proj_b = (const float*)d_in[14];
  const float* lnf_w     = (const float*)d_in[15];
  const float* lnf_b     = (const float*)d_in[16];
  const float* lm_head_w = (const float*)d_in[17];

  char* ws = (char*)d_ws;
  float* xf  = (float*)(ws + 0);          // 4096x768 fp32 residual      (12,582,912 B)
  bf16* hb   = (bf16*)(ws + 12582912);    // 4096x768 bf16 LN output     ( 6,291,456 B)
  bf16* qkvb = (bf16*)(ws + 18874368);    // 4096x2304 bf16              (18,874,368 B)
  bf16* yb   = (bf16*)(ws + 37748736);    // 4096x768 bf16 attn out      ( 6,291,456 B)
  bf16* h2b  = (bf16*)(ws + 44040192);    // 4096x3072 bf16 MLP hidden   (25,165,824 B)

  embed_kernel<<<NROWS, 256, 0, stream>>>(idx, wte, wpe, xf);

  for (int l = 0; l < NLAYER; ++l) {
    ln_kernel<<<NROWS, 256, 0, stream>>>(xf, ln1_w + l * DD, ln1_b + l * DD, hb);
    gemm_kernel<0, false><<<dim3(18, 32), 256, 0, stream>>>(
        hb, c_attn_w + (size_t)l * DD * (3 * DD), c_attn_b + l * (3 * DD),
        qkvb, NROWS, 3 * DD, DD);
    attn_kernel<<<dim3(TT / 16, HH, BB), 32, 0, stream>>>(qkvb, yb);
    gemm_kernel<2, false><<<dim3(6, 32), 256, 0, stream>>>(
        yb, c_proj_w + (size_t)l * DD * DD, c_proj_b + l * DD,
        xf, NROWS, DD, DD);
    ln_kernel<<<NROWS, 256, 0, stream>>>(xf, ln2_w + l * DD, ln2_b + l * DD, hb);
    gemm_kernel<1, false><<<dim3(24, 32), 256, 0, stream>>>(
        hb, fc_w + (size_t)l * DD * (4 * DD), fc_b + l * (4 * DD),
        h2b, NROWS, 4 * DD, DD);
    gemm_kernel<2, false><<<dim3(6, 32), 256, 0, stream>>>(
        h2b, fc_proj_w + (size_t)l * (4 * DD) * DD, fc_proj_b + l * DD,
        xf, NROWS, DD, 4 * DD);
  }

  ln_kernel<<<NROWS, 256, 0, stream>>>(xf, lnf_w, lnf_b, hb);
  gemm_kernel<3, true><<<dim3((VV + 127) / 128, 32), 256, 0, stream>>>(
      hb, lm_head_w, nullptr, d_out, NROWS, VV, DD);
}
